// MPNNGNN_13597866459576
// MI455X (gfx1250) — matrix operand order repaired and checked by reading, not compiled, not run on hardware
//
#include <hip/hip_runtime.h>

typedef __attribute__((ext_vector_type(16))) _Float16 v16h;
typedef __attribute__((ext_vector_type(8)))  _Float16 v8h;
typedef __attribute__((ext_vector_type(8)))  float    v8f;
typedef __attribute__((ext_vector_type(4)))  float    v4f;

#define HID      32
#define NXC      96
#define N_NODES  (6 * NXC * NXC)      // 55296
#define N_EDGES  (4 * N_NODES)        // 221184
#define BATCH    2
#define BN       (BATCH * N_NODES)    // 110592
#define NUM_STEPS 3

// ---------------- WMMA fragment helpers (A/B: contiguous k = 16h+j; C/D per CDNA5 ISA §7.12.2) ----
// A fragment (16x32 f16): lane l holds row m = l&15; vgpr-pair v holds
//   K = 16*(l>=16) + 2*v .. +1
// B fragment (32x16 f16): lane l holds col n = l&15; element j holds K = 16*(l>=16)+j
// C/D (16x16 f32): vgpr v -> row v + 8*(lane>=16), col = lane&15

__device__ __forceinline__ void a_index(int v, int lane, int& m, int& k) {
  m = lane & 15;
  k = ((lane >> 4) << 4) + (v << 1);
}

__device__ __forceinline__ v8f wmma_f16(v16h a, v16h b, v8f c) {
  return __builtin_amdgcn_wmma_f32_16x16x32_f16(false, a, false, b, (short)0, c,
                                                false, false);
}

// Load A fragment from global f32 row-major [rows x 32], base row r0
__device__ __forceinline__ v16h load_a_f32(const float* __restrict__ p, int r0) {
  int lane = threadIdx.x & 31;
  v16h a;
#pragma unroll
  for (int v = 0; v < 8; ++v) {
    int m, k; a_index(v, lane, m, k);
    const float* row = p + (size_t)(r0 + m) * HID;
    a[2 * v]     = (_Float16)row[k];
    a[2 * v + 1] = (_Float16)row[k + 1];
  }
  return a;
}

// Load a pre-packed B fragment (one aligned 32B load -> 2x global_load_b128)
__device__ __forceinline__ v16h load_frag(const _Float16* __restrict__ fragW, int T) {
  int lane = threadIdx.x & 31;
  return *(const v16h*)(fragW + (((T << 5) + lane) << 4));
}

// ---------------- Kernel 0: pack all weight B-fragments into fragment order ----------
// Tiles: T=0..5 -> w_ih^T (gx), 6..11 -> w_hh^T (gh), 12..13 -> pw1, 14..15 -> pw2
// Element (T, lane, j): n = lane&15, k = 16*(lane>=16) + j

__global__ void prep_frag_kernel(const float* __restrict__ wih,
                                 const float* __restrict__ whh,
                                 const float* __restrict__ pw1,
                                 const float* __restrict__ pw2,
                                 _Float16* __restrict__ fragW) {
  int idx = blockIdx.x * 256 + threadIdx.x;      // 8192 elements total
  int T    = idx >> 9;
  int rem  = idx & 511;
  int lane = rem >> 4;
  int j    = rem & 15;
  int n = lane & 15;
  int k = ((lane >> 4) << 4) + j;
  float v;
  if (T < 6)        v = wih[((T * 16 + n) * HID) + k];          // B = w_ih^T
  else if (T < 12)  v = whh[(((T - 6) * 16 + n) * HID) + k];    // B = w_hh^T
  else if (T < 14)  v = pw1[k * HID + (T - 12) * 16 + n];
  else              v = pw2[k * HID + (T - 14) * 16 + n];
  fragW[idx] = (_Float16)v;
}

// ---------------- Kernel 1: node projection h0 = ReLU(x@pw1+pb1)@pw2+pb2 -------------

__global__ void __launch_bounds__(256)
node_proj_kernel(const float* __restrict__ x, const float* __restrict__ pb1,
                 const float* __restrict__ pb2, const _Float16* __restrict__ fragW,
                 float* __restrict__ node, float* __restrict__ hidden) {
  __shared__ float lds[8][16][HID];                      // 16 KB: per-wave repack tile
  int lane = threadIdx.x & 31;
  int wave = threadIdx.x >> 5;
  int tile = blockIdx.x * 8 + wave;
  int r0 = tile * 16;
  int col  = lane & 15;
  int rofs = (lane >> 4) << 3;

  v16h a  = load_a_f32(x, r0);
  v16h b0 = load_frag(fragW, 12);
  v16h b1 = load_frag(fragW, 13);
  v8f z = {};
  v8f c0 = wmma_f16(a, b0, z);
  v8f c1 = wmma_f16(a, b1, z);

#pragma unroll
  for (int v = 0; v < 8; ++v) {
    int r = v + rofs;
    lds[wave][r][col]      = fmaxf(c0[v] + pb1[col], 0.f);
    lds[wave][r][col + 16] = fmaxf(c1[v] + pb1[col + 16], 0.f);
  }
  __syncthreads();

  v16h a2;
#pragma unroll
  for (int v = 0; v < 8; ++v) {
    int m, k; a_index(v, lane, m, k);
    a2[2 * v]     = (_Float16)lds[wave][m][k];
    a2[2 * v + 1] = (_Float16)lds[wave][m][k + 1];
  }
  v16h d0 = load_frag(fragW, 14);
  v16h d1 = load_frag(fragW, 15);
  v8f e0 = wmma_f16(a2, d0, z);
  v8f e1 = wmma_f16(a2, d1, z);

#pragma unroll
  for (int v = 0; v < 8; ++v) {
    size_t r = (size_t)(r0 + v + rofs) * HID;
    float h0v = e0[v] + pb2[col];
    float h1v = e1[v] + pb2[col + 16];
    node[r + col]        = h0v;
    node[r + col + 16]   = h1v;
    hidden[r + col]      = h0v;
    hidden[r + col + 16] = h1v;
  }
}

// ---------------- Kernel 2: edge network We = (ReLU(rel@ew1+eb1))@ew2+eb2 ------------
// Output stored fp16 TRANSPOSED per edge: We_t[e][o][h], so the message kernel
// reads each lane's 32 weights as contiguous 64B (global_load_b128).

__global__ void __launch_bounds__(256)
edge_we_kernel(const float* __restrict__ rel, const float* __restrict__ ew1,
               const float* __restrict__ eb1, const float* __restrict__ ew2,
               const float* __restrict__ eb2, _Float16* __restrict__ We_t) {
  // ew2 staged into LDS already in fragment order: [nt][lane][16 halves]
  __shared__ alignas(32) _Float16 sew2f[64 * 32 * 16];   // 64 KB
  for (int idx = threadIdx.x; idx < 64 * 32 * 16; idx += 256) {
    int nt   = idx >> 9;
    int rem  = idx & 511;
    int l    = rem >> 4;
    int j    = rem & 15;
    int k    = ((l >> 4) << 4) + j;
    int n    = l & 15;
    sew2f[idx] = (_Float16)ew2[(size_t)k * 1024 + nt * 16 + n];
  }
  __syncthreads();

  int lane = threadIdx.x & 31;
  int wave = threadIdx.x >> 5;
  int tile = blockIdx.x * 8 + wave;
  int e0 = tile * 16;
  int col  = lane & 15;
  int rofs = (lane >> 4) << 3;

  // Build A fragment of g = ReLU(rel@ew1+eb1), 16x32, directly in fragment layout
  int m = lane & 15;
  const float* r4 = rel + (size_t)(e0 + m) * 4;
  float q0 = r4[0], q1 = r4[1], q2 = r4[2], q3 = r4[3];
  v16h a;
#pragma unroll
  for (int v = 0; v < 8; ++v) {
    int mm, k; a_index(v, lane, mm, k);
    float g0 = fmaxf(q0 * ew1[0 * HID + k] + q1 * ew1[1 * HID + k] +
                     q2 * ew1[2 * HID + k] + q3 * ew1[3 * HID + k] + eb1[k], 0.f);
    int k1 = k + 1;
    float g1 = fmaxf(q0 * ew1[0 * HID + k1] + q1 * ew1[1 * HID + k1] +
                     q2 * ew1[2 * HID + k1] + q3 * ew1[3 * HID + k1] + eb1[k1], 0.f);
    a[2 * v]     = (_Float16)g0;
    a[2 * v + 1] = (_Float16)g1;
  }

  v8f z = {};
#pragma unroll 4
  for (int nt = 0; nt < 64; ++nt) {
    v16h b = *(const v16h*)&sew2f[((nt << 5) + lane) << 4];   // 2x ds_load_b128
    v8f c = wmma_f16(a, b, z);
    int cbase = nt * 16 + col;          // flattened output column (h*32 + o)
    float bias = eb2[cbase];
    int oo = cbase & 31;
    int hh = cbase >> 5;
#pragma unroll
    for (int v = 0; v < 8; ++v) {
      int r = e0 + v + rofs;
      We_t[(size_t)r * 1024 + oo * 32 + hh] = (_Float16)(c[v] + bias);
    }
  }
}

// ---------------- Kernel 3: zero the aggregation buffer (vectorized) -----------------

__global__ void zero_kernel(v4f* __restrict__ p, int n4) {
  int i = blockIdx.x * blockDim.x + threadIdx.x;
  if (i < n4) p[i] = (v4f){0.f, 0.f, 0.f, 0.f};
}

// ---------------- Kernel 4: message + scatter (one wave per edge, lane = out chan) ---

__global__ void __launch_bounds__(256)
msg_scatter_kernel(const float* __restrict__ node, const _Float16* __restrict__ We_t,
                   const int* __restrict__ esrc, const int* __restrict__ edst,
                   float* __restrict__ agg) {
  int wave = threadIdx.x >> 5;
  int e = blockIdx.x * 8 + wave;
  int o = threadIdx.x & 31;
  int s = esrc[e], d = edst[e];

  const v8h* w = (const v8h*)(We_t + (size_t)e * 1024 + o * 32);   // 64B contiguous
  v8h wv[4];
#pragma unroll
  for (int i = 0; i < 4; ++i) wv[i] = w[i];

  const v4f* n0 = (const v4f*)(node + (size_t)s * HID);
  const v4f* n1 = (const v4f*)(node + ((size_t)N_NODES + s) * HID);
  v4f a0[8], a1[8];
#pragma unroll
  for (int i = 0; i < 8; ++i) { a0[i] = n0[i]; a1[i] = n1[i]; }

  float acc0 = 0.f, acc1 = 0.f;
#pragma unroll
  for (int h = 0; h < HID; ++h) {
    float wf = (float)wv[h >> 3][h & 7];
    acc0 += a0[h >> 2][h & 3] * wf;
    acc1 += a1[h >> 2][h & 3] * wf;
  }
  atomicAdd(&agg[(size_t)d * HID + o], acc0);
  atomicAdd(&agg[((size_t)N_NODES + d) * HID + o], acc1);
}

// ---------------- Kernel 5: ReLU(conv) + GRU step (WMMA for the two gate GEMMs) ------

__global__ void __launch_bounds__(256)
gru_kernel(const float* __restrict__ agg, const float* __restrict__ conv_b,
           const _Float16* __restrict__ fragW,
           const float* __restrict__ b_ih, const float* __restrict__ b_hh,
           float* __restrict__ node, float* __restrict__ hidden) {
  int lane = threadIdx.x & 31;
  int wave = threadIdx.x >> 5;
  int tile = blockIdx.x * 8 + wave;
  int r0 = tile * 16;
  int col  = lane & 15;
  int rofs = (lane >> 4) << 3;

  // node = ReLU(agg + conv_b): build A fragment and persist node for next gather
  v16h a;
#pragma unroll
  for (int v = 0; v < 8; ++v) {
    int m, k; a_index(v, lane, m, k);
    size_t base = (size_t)(r0 + m) * HID;
    float g0 = fmaxf(agg[base + k] + conv_b[k], 0.f);
    float g1 = fmaxf(agg[base + k + 1] + conv_b[k + 1], 0.f);
    node[base + k]     = g0;
    node[base + k + 1] = g1;
    a[2 * v]     = (_Float16)g0;
    a[2 * v + 1] = (_Float16)g1;
  }
  v16h ah = load_a_f32(hidden, r0);

  v8f z = {};
  v8f gx[6], gh[6];
#pragma unroll
  for (int t = 0; t < 6; ++t) {
    gx[t] = wmma_f16(a,  load_frag(fragW, t),     z);
    gh[t] = wmma_f16(ah, load_frag(fragW, 6 + t), z);
  }

#pragma unroll
  for (int t = 0; t < 2; ++t) {
    int cg = t * 16 + col;
    float bir = b_ih[cg],      bhr = b_hh[cg];
    float biz = b_ih[32 + cg], bhz = b_hh[32 + cg];
    float bin = b_ih[64 + cg], bhn = b_hh[64 + cg];
#pragma unroll
    for (int v = 0; v < 8; ++v) {
      size_t idx = (size_t)(r0 + v + rofs) * HID + cg;
      float hprev = hidden[idx];
      float sr = (gx[t][v] + bir) + (gh[t][v] + bhr);
      float sz = (gx[2 + t][v] + biz) + (gh[2 + t][v] + bhz);
      float rg = 1.f / (1.f + __expf(-sr));
      float zg = 1.f / (1.f + __expf(-sz));
      float ng = tanhf((gx[4 + t][v] + bin) + rg * (gh[4 + t][v] + bhn));
      hidden[idx] = (1.f - zg) * ng + zg * hprev;
    }
  }
}

// ---------------- Host launcher ------------------------------------------------------

extern "C" void kernel_launch(void* const* d_in, const int* in_sizes, int n_in,
                              void* d_out, int out_size, void* d_ws, size_t ws_size,
                              hipStream_t stream) {
  (void)in_sizes; (void)n_in; (void)out_size; (void)ws_size;

  const float* x     = (const float*)d_in[0];   // [2,6,96,96,32]
  const float* rel   = (const float*)d_in[1];   // [E,4]
  const float* pw1   = (const float*)d_in[2];
  const float* pb1   = (const float*)d_in[3];
  const float* pw2   = (const float*)d_in[4];
  const float* pb2   = (const float*)d_in[5];
  const float* ew1   = (const float*)d_in[6];
  const float* eb1   = (const float*)d_in[7];
  const float* ew2   = (const float*)d_in[8];   // [32,1024]
  const float* eb2   = (const float*)d_in[9];
  const float* convb = (const float*)d_in[10];
  const float* wih   = (const float*)d_in[11];  // [96,32]
  const float* whh   = (const float*)d_in[12];  // [96,32]
  const float* bih   = (const float*)d_in[13];
  const float* bhh   = (const float*)d_in[14];
  const int*   esrc  = (const int*)d_in[15];
  const int*   edst  = (const int*)d_in[16];

  // Workspace: We_t fp16 (453 MB) + node/hidden/agg fp32 (3 x 13.5 MB) + fragW (16 KB)
  char* ws = (char*)d_ws;
  _Float16* We_t = (_Float16*)ws;
  size_t weBytes = (size_t)N_EDGES * 1024 * sizeof(_Float16);
  float* node   = (float*)(ws + weBytes);
  float* hidden = node + (size_t)BN * HID;
  float* agg    = hidden + (size_t)BN * HID;
  _Float16* fragW = (_Float16*)(agg + (size_t)BN * HID);

  prep_frag_kernel<<<32, 256, 0, stream>>>(wih, whh, pw1, pw2, fragW);
  node_proj_kernel<<<BN / 16 / 8, 256, 0, stream>>>(x, pb1, pb2, fragW, node, hidden);
  edge_we_kernel<<<N_EDGES / 16 / 8, 256, 0, stream>>>(rel, ew1, eb1, ew2, eb2, We_t);

  for (int s = 0; s < NUM_STEPS; ++s) {
    zero_kernel<<<(BN * HID / 4) / 256, 256, 0, stream>>>((v4f*)agg, BN * HID / 4);
    msg_scatter_kernel<<<N_EDGES / 8, 256, 0, stream>>>(node, We_t, esrc, edst, agg);
    gru_kernel<<<BN / 16 / 8, 256, 0, stream>>>(agg, convb, fragW, bih, bhh,
                                                node, hidden);
  }

  hipMemcpyAsync(d_out, hidden, (size_t)BN * HID * sizeof(float),
                 hipMemcpyDeviceToDevice, stream);
}
